// DSSM_PTCR_19585050870116
// MI455X (gfx1250) — matrix operand order repaired and checked by reading, not compiled, not run on hardware
//
#include <hip/hip_runtime.h>
#include <hip/hip_bf16.h>

typedef __attribute__((ext_vector_type(16))) __bf16 v16bf;
typedef __attribute__((ext_vector_type(8)))  float  v8f;

#define ED      64
#define PD      64
#define PH      64
#define HIST    50
#define LFB     200
#define GEN_OUT 8384   // 64 + 4096 + 64 + 4096 + 64
#define HIDL    128

// ---------------------------------------------------------------------------
// Kernel 1: total[b, :] = relu(prompt_item_table[target[b]] @ gen_W + gen_b)
// GEMM  M=B (gathered rows), N=8384, K=64.  bf16 WMMA, f32 accumulate.
// Block = 128 threads (4 waves); block tile = 16 rows x 64 cols.
// grid.x = 8384/64 = 131, grid.y = B/16.
// ---------------------------------------------------------------------------
__global__ __launch_bounds__(128) void gen_gemm_kernel(
    const int*   __restrict__ target_item_id,
    const float* __restrict__ prompt_item_table,
    const float* __restrict__ gen_W,      // 64 x 8384 row-major
    const float* __restrict__ gen_b,      // 8384
    float*       __restrict__ total)      // B x 8384
{
    __shared__ __bf16 sA[16 * 64];
    const int lane  = threadIdx.x & 31;
    const int wave  = threadIdx.x >> 5;
    const int mBase = blockIdx.y * 16;
    const int colBase = blockIdx.x * 64 + wave * 16;

    // Gather 16 rows of A (prompt_item rows), convert to bf16 in LDS.
    for (int i = threadIdx.x; i < 16 * 64; i += 128) {
        int r = i >> 6, c = i & 63;
        int item = target_item_id[mBase + r];
        sA[i] = (__bf16)prompt_item_table[(size_t)item * ED + c];
    }
    __syncthreads();

    // A fragments (16x32 bf16 layout): lane<16 -> K base 0, lane>=16 -> K base 8;
    // elements 0..7 = K+0..7, elements 8..15 = K+16..23.
    const int r  = lane & 15;
    const int kb = (lane >> 4) * 8;
    v16bf a0, a1;
    #pragma unroll
    for (int e = 0; e < 8; ++e) {
        a0[e]     = sA[r * 64 + kb + e];
        a0[8 + e] = sA[r * 64 + kb + 16 + e];
        a1[e]     = sA[r * 64 + 32 + kb + e];
        a1[8 + e] = sA[r * 64 + 32 + kb + 16 + e];
    }

    // B fragments (32x16 bf16): lane<16 -> K base 0, lane>=16 -> K base 8; elements 0..7 = K+0..7, 8..15 = K+16..23.
    const int n   = colBase + (lane & 15);
    const int kb2 = (lane >> 4) * 16;
    v16bf b0, b1;
    #pragma unroll
    for (int e = 0; e < 16; ++e) {
        b0[e] = (__bf16)gen_W[(size_t)(kb + e + (e & 8)) * GEN_OUT + n];
        b1[e] = (__bf16)gen_W[(size_t)(32 + kb + e + (e & 8)) * GEN_OUT + n];
    }

    v8f acc = {};
    acc = __builtin_amdgcn_wmma_f32_16x16x32_bf16(false, a0, false, b0, (short)0, acc, false, false);
    acc = __builtin_amdgcn_wmma_f32_16x16x32_bf16(false, a1, false, b1, (short)0, acc, false, false);

    const float bias = gen_b[n];
    #pragma unroll
    for (int rr = 0; rr < 8; ++rr) {
        int M = rr + ((lane >> 4) << 3);
        float v = acc[rr] + bias;
        v = v > 0.f ? v : 0.f;
        total[(size_t)(mBase + M) * GEN_OUT + n] = v;
    }
}

// ---------------------------------------------------------------------------
// Kernel 2: per-sample 2-layer prompt MLP over 200 pos + 200 neg rows.
//   h   = relu(x @ w1 + b1);  out = h @ w2 + b2;  masked row-sums -> fpos/fneg
//   loss = softplus(-(fpos*neg_num - fneg*pos_num))
// One block (8 waves) per sample; each wave takes 16-row tiles.
// ---------------------------------------------------------------------------
__global__ __launch_bounds__(256) void prompt_mlp_kernel(
    const int*   __restrict__ pos_idx,  const int* __restrict__ pos_mask,
    const int*   __restrict__ neg_idx,  const int* __restrict__ neg_mask,
    const float* __restrict__ prompt_user_table,
    const float* __restrict__ total,
    float*       __restrict__ fpos_out,   // ws: B x 64
    float*       __restrict__ loss_out)   // d_out + B
{
    __shared__ __bf16 sW1[64 * 64];
    __shared__ __bf16 sW2[64 * 64];
    __shared__ float  sB1[64], sB2[64];
    __shared__ float  sAcc[2][64];        // fpos / fneg partials
    __shared__ float  sCnt[2];            // pos_num / neg_num
    __shared__ __bf16 sH[8][16 * 64];     // per-wave hidden re-layout staging
    __shared__ float  sMrow[8][16];       // per-wave row masks

    const int b    = blockIdx.x;
    const int tid  = threadIdx.x;
    const int lane = tid & 31;
    const int wave = tid >> 5;
    const float* row = total + (size_t)b * GEN_OUT;

    // pnet layout within a total row: [0,64)=prompt_embed, [64,4160)=w1,
    // [4160,4224)=b1, [4224,8320)=w2, [8320,8384)=b2
    for (int i = tid; i < 4096; i += 256) {
        sW1[i] = (__bf16)row[64 + i];
        sW2[i] = (__bf16)row[4224 + i];
    }
    if (tid < 64) {
        sB1[tid] = row[4160 + tid];
        sB2[tid] = row[8320 + tid];
        sAcc[0][tid] = 0.f;
        sAcc[1][tid] = 0.f;
    }
    if (tid < 2) sCnt[tid] = 0.f;
    __syncthreads();

    const int r   = lane & 15;
    const int kb  = (lane >> 4) * 8;
    const int kb2 = (lane >> 4) * 16;
    const int cn  = lane & 15;
    const int mh  = (lane >> 4) << 3;     // 0 or 8: row-half for C layout

    // 13 tiles pos (t=0..12) + 13 tiles neg (t=13..25)
    for (int t = wave; t < 26; t += 8) {
        const int  branch = (t < 13) ? 0 : 1;
        const int  tb     = (branch == 0 ? t : t - 13) * 16;
        const int* idxA   = branch == 0 ? pos_idx  : neg_idx;
        const int* mskA   = branch == 0 ? pos_mask : neg_mask;

        const int  g     = tb + r;
        const bool valid = (g < LFB);
        const int  uidx  = valid ? idxA[b * LFB + g] : 0;
        const float m    = valid ? (float)mskA[b * LFB + g] : 0.f;
        sMrow[wave][r] = m;                       // lanes l and l+16 write same value
        if (lane < 16) atomicAdd(&sCnt[branch], m);

        // A fragments: gathered prompt_user row (64 dims) -> 2 bf16 frags
        const float* ue = prompt_user_table + (size_t)uidx * ED;
        v16bf a0, a1;
        #pragma unroll
        for (int e = 0; e < 8; ++e) {
            a0[e]     = (__bf16)ue[kb + e];
            a0[8 + e] = (__bf16)ue[kb + 16 + e];
            a1[e]     = (__bf16)ue[32 + kb + e];
            a1[8 + e] = (__bf16)ue[32 + kb + 16 + e];
        }

        // ---- layer 1: H(16x64) = relu(A @ w1 + b1), 4 column tiles ----
        #pragma unroll
        for (int nt = 0; nt < 4; ++nt) {
            const int c = nt * 16 + cn;
            v16bf w0, w1f;
            #pragma unroll
            for (int e = 0; e < 16; ++e) {
                w0[e]  = sW1[(kb + e + (e & 8)) * 64 + c];
                w1f[e] = sW1[(32 + kb + e + (e & 8)) * 64 + c];
            }
            v8f h = {};
            h = __builtin_amdgcn_wmma_f32_16x16x32_bf16(false, a0, false, w0,  (short)0, h, false, false);
            h = __builtin_amdgcn_wmma_f32_16x16x32_bf16(false, a1, false, w1f, (short)0, h, false, false);
            const float b1v = sB1[c];
            #pragma unroll
            for (int rr = 0; rr < 8; ++rr) {
                float v = h[rr] + b1v;
                v = v > 0.f ? v : 0.f;
                sH[wave][(rr + mh) * 64 + c] = (__bf16)v;   // intra-wave LDS, ordered
            }
        }

        // Re-layout hidden as A fragments (intra-wave LDS round trip)
        v16bf ha0, ha1;
        #pragma unroll
        for (int e = 0; e < 8; ++e) {
            ha0[e]     = sH[wave][r * 64 + kb + e];
            ha0[8 + e] = sH[wave][r * 64 + kb + 16 + e];
            ha1[e]     = sH[wave][r * 64 + 32 + kb + e];
            ha1[8 + e] = sH[wave][r * 64 + 32 + kb + 16 + e];
        }

        // ---- layer 2: OUT = H @ w2 + b2; masked row-sum into sAcc ----
        #pragma unroll
        for (int nt = 0; nt < 4; ++nt) {
            const int c = nt * 16 + cn;
            v16bf w0, w1f;
            #pragma unroll
            for (int e = 0; e < 16; ++e) {
                w0[e]  = sW2[(kb + e + (e & 8)) * 64 + c];
                w1f[e] = sW2[(32 + kb + e + (e & 8)) * 64 + c];
            }
            v8f o = {};
            o = __builtin_amdgcn_wmma_f32_16x16x32_bf16(false, ha0, false, w0,  (short)0, o, false, false);
            o = __builtin_amdgcn_wmma_f32_16x16x32_bf16(false, ha1, false, w1f, (short)0, o, false, false);
            const float b2v = sB2[c];
            float s = 0.f;
            #pragma unroll
            for (int rr = 0; rr < 8; ++rr)
                s += (o[rr] + b2v) * sMrow[wave][rr + mh];
            atomicAdd(&sAcc[branch][c], s);     // ds_add_f32
        }
    }
    __syncthreads();

    if (tid < 64) {
        const float fp = sAcc[0][tid], fn = sAcc[1][tid];
        const float pn = sCnt[0], nn = sCnt[1];
        const float z  = -(fp * nn - fn * pn);
        const float loss = fmaxf(z, 0.f) + log1pf(expf(-fabsf(z)));   // stable softplus
        fpos_out[(size_t)b * 64 + tid] = fp;
        loss_out[(size_t)b * 64 + tid] = loss;
    }
}

// ---------------------------------------------------------------------------
// Kernel 3: history pooling, user/item towers, fusion, final dot product.
// Tiny FLOPs (~0.5 GFLOP total) -> plain VALU, one block (128 thr) per sample.
// ---------------------------------------------------------------------------
__global__ __launch_bounds__(128) void tower_fuse_kernel(
    const int*   __restrict__ user_id,
    const int*   __restrict__ target_item_id,
    const int*   __restrict__ history_item_id,
    const int*   __restrict__ history_len,
    const float* __restrict__ user_features,
    const float* __restrict__ item_features,
    const float* __restrict__ user_emb_table,
    const float* __restrict__ item_emb_table,
    const float* __restrict__ uW1, const float* __restrict__ ub1,
    const float* __restrict__ uW2, const float* __restrict__ ub2,
    const float* __restrict__ iW1, const float* __restrict__ ib1,
    const float* __restrict__ iW2, const float* __restrict__ ib2,
    const float* __restrict__ fusion_W,
    const float* __restrict__ total,
    const float* __restrict__ fpos,
    float*       __restrict__ out)     // d_out[0..B)
{
    __shared__ float xu[160];     // [ue, pooled, user_features]
    __shared__ float xi[96];      // [item_emb, item_features]
    __shared__ float h[128];
    __shared__ float fu[64];
    __shared__ float cat[192];    // [final_item, fpos, prompt_embed]
    __shared__ float prod[64];

    const int b = blockIdx.x, tid = threadIdx.x;
    const int uid = user_id[b], iid = target_item_id[b];
    const int hl  = history_len[b];

    if (tid < 64) {
        xu[tid] = user_emb_table[(size_t)uid * ED + tid];
        float s = 0.f;
        for (int t = 0; t < HIST; ++t)
            if (t < hl)
                s += item_emb_table[(size_t)history_item_id[b * HIST + t] * ED + tid];
        xu[64 + tid] = s / (float)(hl > 0 ? hl : 1);
        xi[tid] = item_emb_table[(size_t)iid * ED + tid];
    } else if (tid < 96) {
        const int j = tid - 64;
        xu[128 + j] = user_features[b * 32 + j];
        xi[64 + j]  = item_features[b * 32 + j];
    }
    __syncthreads();

    // user tower hidden (160 -> 128)
    {
        float s = ub1[tid];
        for (int d = 0; d < 160; ++d) s += xu[d] * uW1[d * HIDL + tid];
        h[tid] = fmaxf(s, 0.f);
    }
    __syncthreads();
    if (tid < 64) {
        float s = ub2[tid];
        for (int d = 0; d < HIDL; ++d) s += h[d] * uW2[d * 64 + tid];
        fu[tid] = fmaxf(s, 0.f);
    }
    __syncthreads();

    // item tower hidden (96 -> 128), reuse h
    {
        float s = ib1[tid];
        for (int d = 0; d < 96; ++d) s += xi[d] * iW1[d * HIDL + tid];
        h[tid] = fmaxf(s, 0.f);
    }
    __syncthreads();
    if (tid < 64) {
        float s = ib2[tid];
        for (int d = 0; d < HIDL; ++d) s += h[d] * iW2[d * 64 + tid];
        cat[tid]       = fmaxf(s, 0.f);                       // final_item
        cat[64 + tid]  = fpos[(size_t)b * 64 + tid];          // fpos
        cat[128 + tid] = total[(size_t)b * GEN_OUT + tid];    // prompt_embed
    }
    __syncthreads();

    if (tid < 64) {
        float s = 0.f;
        for (int d = 0; d < 192; ++d) s += cat[d] * fusion_W[d * 64 + tid];
        prod[tid] = fu[tid] * fmaxf(s, 0.f);
    }
    __syncthreads();
    if (tid == 0) {
        float s = 0.f;
        for (int j = 0; j < 64; ++j) s += prod[j];
        out[b] = s;
    }
}

// ---------------------------------------------------------------------------
extern "C" void kernel_launch(void* const* d_in, const int* in_sizes, int n_in,
                              void* d_out, int out_size, void* d_ws, size_t ws_size,
                              hipStream_t stream) {
    const int*   user_id        = (const int*)  d_in[0];
    const int*   target_item_id = (const int*)  d_in[1];
    const int*   history_items  = (const int*)  d_in[2];
    const int*   history_len    = (const int*)  d_in[3];
    const float* user_features  = (const float*)d_in[4];
    const float* item_features  = (const float*)d_in[5];
    const int*   pos_idx        = (const int*)  d_in[6];
    const int*   pos_mask       = (const int*)  d_in[7];
    const int*   neg_idx        = (const int*)  d_in[8];
    const int*   neg_mask       = (const int*)  d_in[9];
    const float* user_emb       = (const float*)d_in[10];
    const float* item_emb       = (const float*)d_in[11];
    const float* prompt_item    = (const float*)d_in[12];
    const float* prompt_user    = (const float*)d_in[13];
    const float* uW1            = (const float*)d_in[14];
    const float* ub1            = (const float*)d_in[15];
    const float* uW2            = (const float*)d_in[16];
    const float* ub2            = (const float*)d_in[17];
    const float* iW1            = (const float*)d_in[18];
    const float* ib1            = (const float*)d_in[19];
    const float* iW2            = (const float*)d_in[20];
    const float* ib2            = (const float*)d_in[21];
    const float* gen_W          = (const float*)d_in[22];
    const float* gen_b          = (const float*)d_in[23];
    const float* fusion_W       = (const float*)d_in[24];

    const int B = in_sizes[0];                 // 4096

    float* total = (float*)d_ws;               // B x 8384
    float* fpos  = total + (size_t)B * GEN_OUT; // B x 64
    float* out   = (float*)d_out;              // [0,B): output
    float* loss  = out + B;                    // [B, B+64B): loss_pfpe

    dim3 g1(GEN_OUT / 64, B / 16);
    gen_gemm_kernel<<<g1, 128, 0, stream>>>(target_item_id, prompt_item, gen_W, gen_b, total);

    prompt_mlp_kernel<<<B, 256, 0, stream>>>(pos_idx, pos_mask, neg_idx, neg_mask,
                                             prompt_user, total, fpos, loss);

    tower_fuse_kernel<<<B, 128, 0, stream>>>(user_id, target_item_id, history_items,
                                             history_len, user_features, item_features,
                                             user_emb, item_emb,
                                             uW1, ub1, uW2, ub2,
                                             iW1, ib1, iW2, ib2,
                                             fusion_W, total, fpos, out);
}
